// GRUD_cell_22187801051697
// MI455X (gfx1250) — matrix operand order repaired and checked by reading, not compiled, not run on hardware
//
#include <hip/hip_runtime.h>
#include <hip/hip_bf16.h>
#include <math.h>

// ---------------- problem constants ----------------
#define Bn 256
#define Cn 64
#define Tn 1000
#define Hn 64
#define On 32
#define Mrows (Tn * Bn)   // 256000 rows for the batched GEMMs

typedef __attribute__((ext_vector_type(16))) __bf16 v16bf;
typedef __attribute__((ext_vector_type(8)))  float  v8f;

// Fragment <-> 2x16B views for vectorized loads
union Frag { v16bf bf; uint4 q[2]; };

// fast device math on the recurrent critical path: hardware v_rcp_f32
// instead of the IEEE division expansion (bf16 WMMA inputs dominate the
// error budget anyway)
__device__ inline float fast_sigmoid(float x) {
    return __builtin_amdgcn_rcpf(1.f + __expf(-x));
}
__device__ inline float fast_tanh(float x) {
    const float e = __expf(-2.f * x);
    return (1.f - e) * __builtin_amdgcn_rcpf(1.f + e);
}

// =====================================================================
// Weight prep: emit bf16 weights directly in WMMA B-fragment-packed
// layout.  For a conceptual Bt[K][N] (= W^T), tile (kt,nt) fragment is
// stored as 32 lanes x 16 bf16 contiguous:
//   p = (((nt*KT + kt)*32 + lane)*16 + e)
//   k = kt*32 + (lane>>4)*16 + e ;  n = nt*16 + (lane&15)
// (contiguous k order chosen here, same as the A fragments: lanes 0-15 hold K 0..15,
//  lanes 16-31 hold K 16..31, N = lane%16)
//   Wdg_pk   K=64  N=128 (gamma_x | gamma_h)
//   Wxm_pk   K=128 N=192 (x2-part rows 0..63, m-part rows 64..127)
//   Whh_pk   K=64  N=224 (W_hz|W_hr|W_hh|W_hy)^T
// =====================================================================
__global__ __launch_bounds__(256) void k_prep_weights(
    const float* __restrict__ W_dgx, const float* __restrict__ b_dgx,
    const float* __restrict__ W_dgh, const float* __restrict__ b_dgh,
    const float* __restrict__ W_xz,  const float* __restrict__ W_hz,
    const float* __restrict__ W_mz,  const float* __restrict__ b_mz,
    const float* __restrict__ W_xr,  const float* __restrict__ W_hr,
    const float* __restrict__ W_mr,
    const float* __restrict__ W_xh,  const float* __restrict__ W_hh,
    const float* __restrict__ W_mh,  const float* __restrict__ b_mh,
    const float* __restrict__ W_hy,
    __bf16* __restrict__ Wdg_pk, __bf16* __restrict__ Wxm_pk,
    __bf16* __restrict__ Whh_pk,
    float* __restrict__ bias_dg, float* __restrict__ bias3)
{
    const int tid = threadIdx.x;

    // ---- Wdg_pk: K=64 (KT=2), N=128 (NT=8) ----
    for (int p = tid; p < 64 * 128; p += 256) {
        const int e = p & 15, lane = (p >> 4) & 31, q = p >> 9;
        const int kt = q & 1, nt = q >> 1;                 // KT=2
        const int k = kt * 32 + ((lane >> 4) << 4) + e;
        const int n = nt * 16 + (lane & 15);
        const float v = (n < 64) ? W_dgx[n * 64 + k] : W_dgh[(n - 64) * 64 + k];
        Wdg_pk[p] = (__bf16)v;
    }
    for (int i = tid; i < 128; i += 256)
        bias_dg[i] = (i < 64) ? b_dgx[i] : b_dgh[i - 64];

    // ---- Wxm_pk: K=128 (KT=4), N=192 (NT=12) ----
    for (int p = tid; p < 128 * 192; p += 256) {
        const int e = p & 15, lane = (p >> 4) & 31, q = p >> 9;
        const int kt = q & 3, nt = q >> 2;                 // KT=4
        const int k = kt * 32 + ((lane >> 4) << 4) + e;
        const int n = nt * 16 + (lane & 15);
        const int nn = n & 63;
        const float* Wx = (n < 64) ? W_xz : (n < 128) ? W_xr : W_xh;
        const float* Wm = (n < 64) ? W_mz : (n < 128) ? W_mr : W_mh;
        const float v = (k < 64) ? Wx[nn * 64 + k] : Wm[nn * 64 + (k - 64)];
        Wxm_pk[p] = (__bf16)v;
    }
    for (int i = tid; i < 192; i += 256)
        bias3[i] = (i < 64) ? b_mz[i] : (i < 128) ? 0.f : b_mh[i - 128];

    // ---- Whh_pk: K=64 (KT=2), N=224 (NT=14) ----
    for (int p = tid; p < 64 * 224; p += 256) {
        const int e = p & 15, lane = (p >> 4) & 31, q = p >> 9;
        const int kt = q & 1, nt = q >> 1;                 // KT=2
        const int k = kt * 32 + ((lane >> 4) << 4) + e;
        const int n = nt * 16 + (lane & 15);
        float v;
        if      (n < 64)  v = W_hz[n * 64 + k];
        else if (n < 128) v = W_hr[(n - 64) * 64 + k];
        else if (n < 192) v = W_hh[(n - 128) * 64 + k];
        else              v = W_hy[(n - 192) * 64 + k];
        Whh_pk[p] = (__bf16)v;
    }
}

// =====================================================================
// Coalesced transpose Delta [B,C,T] -> time-major bf16 [T*B, C]
// via a 64x64 LDS tile (reads 256B rows, writes 128B rows).
// =====================================================================
__global__ __launch_bounds__(256) void k_pack_delta(
    const float* __restrict__ Delta, __bf16* __restrict__ Dsbf)
{
    __shared__ float tile[64][65];
    const int b  = blockIdx.x;        // batch row
    const int t0 = blockIdx.y * 64;   // time tile
    for (int i = threadIdx.x; i < 64 * 64; i += 256) {
        const int c = i >> 6, t = i & 63;
        tile[c][t] = (t0 + t < Tn)
                   ? Delta[((size_t)b * Cn + c) * Tn + t0 + t] : 0.f;
    }
    __syncthreads();
    for (int i = threadIdx.x; i < 64 * 64; i += 256) {
        const int t = i >> 6, c = i & 63;
        if (t0 + t < Tn)
            Dsbf[((size_t)(t0 + t) * Bn + b) * Cn + c] = (__bf16)tile[c][t];
    }
}

// =====================================================================
// Single-wave row-block WMMA GEMM:
//   C[m0:m0+16, :] = act( A[M, KT*32] @ Bt + bias )
// A row-major bf16 (each fragment = two global_load_b128), B pre-packed
// in fragment layout (two b128 from L2).  Each A fragment is loaded
// exactly once and re-used across all NT output tiles held in VGPRs.
// act: 0 = identity, 1 = exp(-relu(x))  (GRU-D decay gate)
// =====================================================================
template <int KT, int NT, int ACT>
__global__ __launch_bounds__(32) void k_wmma_gemm(
    const __bf16* __restrict__ A, const __bf16* __restrict__ Bpk,
    const float* __restrict__ bias, float* __restrict__ Cout)
{
    const int lane = threadIdx.x & 31;
    const int lhi = lane >> 4, llo = lane & 15;
    const int m0 = blockIdx.x * 16;
    constexpr int Kd = KT * 32, Nc = NT * 16;

    v8f acc[NT];
#pragma unroll
    for (int nt = 0; nt < NT; ++nt) {
        const float bv = bias[nt * 16 + llo];
#pragma unroll
        for (int v = 0; v < 8; ++v) acc[nt][v] = bv;
    }

#pragma unroll
    for (int kt = 0; kt < KT; ++kt) {
        Frag fa;   // A-frag: VGPR0-7 = K lhi*16+0..15
        const __bf16* arow = A + (size_t)(m0 + llo) * Kd + kt * 32 + lhi * 16;
        fa.q[0] = *(const uint4*)(arow);
        fa.q[1] = *(const uint4*)(arow + 8);
#pragma unroll
        for (int nt = 0; nt < NT; ++nt) {
            Frag fb;
            const __bf16* bp = Bpk + (((size_t)(nt * KT + kt) * 32 + lane) << 4);
            fb.q[0] = *(const uint4*)(bp);
            fb.q[1] = *(const uint4*)(bp + 8);
            acc[nt] = __builtin_amdgcn_wmma_f32_16x16x32_bf16(
                false, fa.bf, false, fb.bf, (short)0, acc[nt], false, false);
        }
    }
#pragma unroll
    for (int nt = 0; nt < NT; ++nt)
#pragma unroll
        for (int v = 0; v < 8; ++v) {
            float x = acc[nt][v];
            if (ACT == 1) x = __expf(-fmaxf(x, 0.f));
            Cout[(size_t)(m0 + v + (lhi << 3)) * Nc + nt * 16 + llo] = x;
        }
}

// =====================================================================
// Per-(b,c) sequential imputation scan over t. Each thread streams
// contiguous X/Mask addresses over its t-loop; gx reads and X2M writes
// are lane-coalesced.  Produces packed bf16 rows [x2 | m] (width 128).
// =====================================================================
__global__ __launch_bounds__(256) void k_scan_impute(
    const float* __restrict__ X, const float* __restrict__ Mask,
    const float* __restrict__ x_mean, const float* __restrict__ Gbuf,
    __bf16* __restrict__ X2M)
{
    const int tid = blockIdx.x * 256 + threadIdx.x;   // 0 .. B*C-1
    if (tid >= Bn * Cn) return;
    const int b = tid >> 6, c = tid & 63;
    const size_t src = ((size_t)b * Cn + c) * Tn;
    const float xm = x_mean[c];
    float xlast = 0.f;
    for (int t = 0; t < Tn; ++t) {
        const float x = X[src + t];
        const float m = Mask[src + t];
        const float gx = Gbuf[((size_t)t * Bn + b) * 128 + c];
        if (m > 0.f) xlast = x;
        const float x1 = m * x + (1.f - m) * (gx * x     + (1.f - gx) * xm);
        const float x2 = m * x1 + (1.f - m) * (gx * xlast + (1.f - gx) * xm);
        const size_t o = ((size_t)t * Bn + b) * 128;
        X2M[o + c]      = (__bf16)x2;
        X2M[o + 64 + c] = (__bf16)m;
    }
}

// =====================================================================
// Recurrent kernel. 16 blocks x 128 threads (4 waves); block owns 16
// batch rows, h[16][64] in LDS, wave j owns hidden cols 16j..16j+15 and
// keeps its W_hz/W_hr/W_hh/W_hy fragments in VGPRs across all T steps.
// =====================================================================
#define LDW 68   // LDS row stride (floats); 272B rows stay 16B-aligned

// A-fragment (16-bit 16x32) from an LDS fp32 row: two contiguous
// 8-float runs -> vectorized ds loads + v_cvt_pk_bf16_f32.
__device__ inline v16bf lds_afrag(const float* __restrict__ row,
                                  int kc, int lhi)
{
    const float* p = row + kc * 32 + (lhi << 4);
    v16bf r;
#pragma unroll
    for (int i = 0; i < 8; ++i) r[i] = (__bf16)p[i];
#pragma unroll
    for (int i = 0; i < 8; ++i) r[8 + i] = (__bf16)p[8 + i];
    return r;
}

__global__ __launch_bounds__(128) void k_recurrent(
    const float* __restrict__ Gbuf, const float* __restrict__ A3,
    const __bf16* __restrict__ Whh_pk, const float* __restrict__ b_hy,
    float* __restrict__ outs, float* __restrict__ hs)
{
    __shared__ float h_lds[16][LDW];
    __shared__ float rh_lds[16][LDW];

    const int lane = threadIdx.x & 31;
    const int wv = threadIdx.x >> 5;
    const int lhi = lane >> 4, llo = lane & 15;
    const int b0 = blockIdx.x * 16;

    for (int i = threadIdx.x; i < 16 * LDW; i += 128)
        (&h_lds[0][0])[i] = 0.f;

    // persistent weight fragments from packed layout (KT=2, NT=14)
    v16bf fz[2], fr[2], fh[2], fy[2];
#pragma unroll
    for (int kc = 0; kc < 2; ++kc) {
        Frag f;
        const __bf16* base = Whh_pk + ((size_t)lane << 4) + (kc << 9);
        // nt stride = KT*32*16 = 1024 elements
        f.q[0] = *(const uint4*)(base + (size_t)wv * 1024);
        f.q[1] = *(const uint4*)(base + (size_t)wv * 1024 + 8);
        fz[kc] = f.bf;
        f.q[0] = *(const uint4*)(base + (size_t)(4 + wv) * 1024);
        f.q[1] = *(const uint4*)(base + (size_t)(4 + wv) * 1024 + 8);
        fr[kc] = f.bf;
        f.q[0] = *(const uint4*)(base + (size_t)(8 + wv) * 1024);
        f.q[1] = *(const uint4*)(base + (size_t)(8 + wv) * 1024 + 8);
        fh[kc] = f.bf;
        f.q[0] = *(const uint4*)(base + (size_t)(12 + (wv & 1)) * 1024);
        f.q[1] = *(const uint4*)(base + (size_t)(12 + (wv & 1)) * 1024 + 8);
        fy[kc] = f.bf;
    }
    const float by = b_hy[(wv & 1) * 16 + llo];
    __syncthreads();

    for (int t = 0; t < Tn; ++t) {
        const size_t rowbase = (size_t)t * Bn + b0;

        // prefetch next step's A3 block (global_prefetch_b8)
        if (wv == 0 && t + 1 < Tn) {
            const char* nxt = (const char*)(A3 + ((size_t)(t + 1) * Bn + b0) * 192);
            for (int i = lane; i < 48; i += 32)
                __builtin_prefetch(nxt + i * 256, 0, 1);
        }

        // ---- phase 1: decay h (wave wv owns its column block) ----
        for (int i = lane; i < 256; i += 32) {
            const int rM = i >> 4, cN = wv * 16 + (i & 15);
            h_lds[rM][cN] *= Gbuf[(rowbase + rM) * 128 + 64 + cN];
        }
        __syncthreads();

        // ---- phase 2: z and r gates (C init from precomputed A3) ----
        v16bf ha0 = lds_afrag(&h_lds[llo][0], 0, lhi);
        v16bf ha1 = lds_afrag(&h_lds[llo][0], 1, lhi);
        v8f az, ar;
#pragma unroll
        for (int v = 0; v < 8; ++v) {
            const int rM = v + (lhi << 3);
            az[v] = A3[(rowbase + rM) * 192 +      wv * 16 + llo];
            ar[v] = A3[(rowbase + rM) * 192 + 64 + wv * 16 + llo];
        }
        az = __builtin_amdgcn_wmma_f32_16x16x32_bf16(false, ha0, false, fz[0], (short)0, az, false, false);
        az = __builtin_amdgcn_wmma_f32_16x16x32_bf16(false, ha1, false, fz[1], (short)0, az, false, false);
        ar = __builtin_amdgcn_wmma_f32_16x16x32_bf16(false, ha0, false, fr[0], (short)0, ar, false, false);
        ar = __builtin_amdgcn_wmma_f32_16x16x32_bf16(false, ha1, false, fr[1], (short)0, ar, false, false);
#pragma unroll
        for (int v = 0; v < 8; ++v) {
            az[v] = fast_sigmoid(az[v]);
            ar[v] = fast_sigmoid(ar[v]);
        }

        // ---- r*h into LDS ----
#pragma unroll
        for (int v = 0; v < 8; ++v) {
            const int rM = v + (lhi << 3), cN = wv * 16 + llo;
            rh_lds[rM][cN] = ar[v] * h_lds[rM][cN];
        }
        __syncthreads();

        // ---- h_tilde = tanh(A3_h + (r*h) @ Whh^T) ----
        v16bf ra0 = lds_afrag(&rh_lds[llo][0], 0, lhi);
        v16bf ra1 = lds_afrag(&rh_lds[llo][0], 1, lhi);
        v8f ah;
#pragma unroll
        for (int v = 0; v < 8; ++v)
            ah[v] = A3[(rowbase + v + (lhi << 3)) * 192 + 128 + wv * 16 + llo];
        ah = __builtin_amdgcn_wmma_f32_16x16x32_bf16(false, ra0, false, fh[0], (short)0, ah, false, false);
        ah = __builtin_amdgcn_wmma_f32_16x16x32_bf16(false, ra1, false, fh[1], (short)0, ah, false, false);

        // ---- h update + store hs (own column block only) ----
#pragma unroll
        for (int v = 0; v < 8; ++v) {
            const int rM = v + (lhi << 3), cN = wv * 16 + llo;
            const float ht = fast_tanh(ah[v]);
            const float hn = (1.f - az[v]) * h_lds[rM][cN] + az[v] * ht;
            h_lds[rM][cN] = hn;
            hs[((size_t)(b0 + rM) * Tn + t) * Hn + cN] = hn;
        }
        __syncthreads();

        // ---- output head: waves 0,1 compute sig(h @ Why + b_hy) ----
        if (wv < 2) {
            v16bf hn0 = lds_afrag(&h_lds[llo][0], 0, lhi);
            v16bf hn1 = lds_afrag(&h_lds[llo][0], 1, lhi);
            v8f ao;
#pragma unroll
            for (int v = 0; v < 8; ++v) ao[v] = by;
            ao = __builtin_amdgcn_wmma_f32_16x16x32_bf16(false, hn0, false, fy[0], (short)0, ao, false, false);
            ao = __builtin_amdgcn_wmma_f32_16x16x32_bf16(false, hn1, false, fy[1], (short)0, ao, false, false);
#pragma unroll
            for (int v = 0; v < 8; ++v) {
                const int rM = v + (lhi << 3), o = wv * 16 + llo;
                outs[((size_t)(b0 + rM) * Tn + t) * On + o] = fast_sigmoid(ao[v]);
            }
        }
        __syncthreads();
    }
}

// =====================================================================
// Launch
// =====================================================================
extern "C" void kernel_launch(void* const* d_in, const int* in_sizes, int n_in,
                              void* d_out, int out_size, void* d_ws, size_t ws_size,
                              hipStream_t stream)
{
    (void)in_sizes; (void)n_in; (void)out_size; (void)ws_size;
    const float* X      = (const float*)d_in[0];
    const float* Mask   = (const float*)d_in[1];
    const float* Delta  = (const float*)d_in[2];
    const float* x_mean = (const float*)d_in[3];
    const float* W_dgx  = (const float*)d_in[4];
    const float* b_dgx  = (const float*)d_in[5];
    const float* W_dgh  = (const float*)d_in[6];
    const float* b_dgh  = (const float*)d_in[7];
    const float* W_xz   = (const float*)d_in[8];
    const float* W_hz   = (const float*)d_in[9];
    const float* W_mz   = (const float*)d_in[10];
    const float* b_mz   = (const float*)d_in[11];
    const float* W_xr   = (const float*)d_in[12];
    const float* W_hr   = (const float*)d_in[13];
    const float* W_mr   = (const float*)d_in[14];
    const float* W_xh   = (const float*)d_in[15];
    const float* W_hh   = (const float*)d_in[16];
    const float* W_mh   = (const float*)d_in[17];
    const float* b_mh   = (const float*)d_in[18];
    const float* W_hy   = (const float*)d_in[19];
    const float* b_hy   = (const float*)d_in[20];

    // ---- workspace carve-up (256B aligned) ----
    char* ws = (char*)d_ws;
    size_t off = 0;
    auto carve = [&](size_t bytes) { char* p = ws + off;
        off += (bytes + 255) & ~(size_t)255; return p; };
    __bf16* Dsbf   = (__bf16*)carve((size_t)Mrows * Cn * sizeof(__bf16));
    float*  Gbuf   = (float*) carve((size_t)Mrows * 128 * sizeof(float));
    __bf16* X2M    = (__bf16*)carve((size_t)Mrows * 128 * sizeof(__bf16));
    float*  A3     = (float*) carve((size_t)Mrows * 192 * sizeof(float));
    __bf16* Wdg_pk = (__bf16*)carve(64 * 128 * sizeof(__bf16));
    __bf16* Wxm_pk = (__bf16*)carve(128 * 192 * sizeof(__bf16));
    __bf16* Whh_pk = (__bf16*)carve(64 * 224 * sizeof(__bf16));
    float*  biasdg = (float*) carve(128 * sizeof(float));
    float*  bias3  = (float*) carve(192 * sizeof(float));

    float* outs = (float*)d_out;                        // [B,T,O]
    float* hs   = (float*)d_out + (size_t)Bn * Tn * On; // [B,T,H]

    // 1. weight prep (fragment-packed bf16 + fused biases)
    k_prep_weights<<<1, 256, 0, stream>>>(
        W_dgx, b_dgx, W_dgh, b_dgh, W_xz, W_hz, W_mz, b_mz,
        W_xr, W_hr, W_mr, W_xh, W_hh, W_mh, b_mh, W_hy,
        Wdg_pk, Wxm_pk, Whh_pk, biasdg, bias3);

    // 2. coalesced transpose Delta -> time-major bf16
    k_pack_delta<<<dim3(Bn, (Tn + 63) / 64), 256, 0, stream>>>(Delta, Dsbf);

    // 3. gammas: G = exp(-relu(D @ Wdg + bias))   [Mrows, 128]
    k_wmma_gemm<2, 8, 1><<<Mrows / 16, 32, 0, stream>>>(
        Dsbf, Wdg_pk, biasdg, Gbuf);

    // 4. imputation scan -> packed [x2 | m] bf16  [Mrows, 128]
    k_scan_impute<<<(Bn * Cn) / 256, 256, 0, stream>>>(
        X, Mask, x_mean, Gbuf, X2M);

    // 5. gate pre-activations A3 = [x2|m] @ Wxm + bias3   [Mrows, 192]
    k_wmma_gemm<4, 12, 0><<<Mrows / 16, 32, 0, stream>>>(
        X2M, Wxm_pk, bias3, A3);

    // 6. sequential recurrence (16 independent batch blocks)
    k_recurrent<<<16, 128, 0, stream>>>(Gbuf, A3, Whh_pk, b_hy, outs, hs);
}
